// Tacotron2_27599459844834
// MI455X (gfx1250) — matrix operand order repaired and checked by reading, not compiled, not run on hardware
//
#include <hip/hip_runtime.h>
#include <hip/hip_bf16.h>

typedef __bf16 bf16;
typedef __attribute__((ext_vector_type(16))) __bf16 v16bf;
typedef __attribute__((ext_vector_type(8)))  float  v8f;

#define B16   16     // batch padded to one WMMA M-tile (real B = 8)
#define SLEN  128
#define HDIM  512
#define TSTEPS 200
#define NMELS 80

// ---------------- small helpers ----------------
__device__ inline bf16 f2bf(float f) {
  unsigned u = __float_as_uint(f);
  unsigned r = (u + 0x7FFFu + ((u >> 16) & 1u)) >> 16;
  unsigned short s = (unsigned short)r;
  bf16 b; __builtin_memcpy(&b, &s, 2);
  return b;
}
__device__ inline float sigm(float x) { return 1.0f / (1.0f + __expf(-x)); }
__device__ inline v8f vzero() { v8f v = {0.f,0.f,0.f,0.f,0.f,0.f,0.f,0.f}; return v; }

// One 16x16 f32 tile of A(16xK) @ W.T where W is row-major [N][K] (bf16).
// A row-major [16][lda] bf16. K must be a multiple of 32.
// Fragment layouts per CDNA5 ISA 7.12.2 (wave32):
//   A: lane(m=lane&15, half=lane>>4), elem e -> K = (e&7) + 8*half + 16*(e>>3)
//   B: lane(n=lane&15, half),          elem e -> K = (e&7) + 8*half + 16*(e>>3)
__device__ inline v8f wmma_tile(const bf16* __restrict__ A, int lda,
                                const bf16* __restrict__ Bt, int ldb,
                                int K, v8f acc) {
  const int lane = threadIdx.x & 31;
  const int half = lane >> 4;
  const int mn = lane & 15;
  for (int kb = 0; kb < K; kb += 32) {
    v16bf a, b;
#pragma unroll
    for (int e = 0; e < 16; ++e) {
      int ka = kb + (e & 7) + (half << 3) + ((e >> 3) << 4);
      int kn = kb + (e & 7) + (half << 3) + ((e >> 3) << 4);
      a[e] = A[mn * lda + ka];
      b[e] = Bt[mn * ldb + kn];
    }
    acc = __builtin_amdgcn_wmma_f32_16x16x32_bf16(false, a, false, b,
                                                  (short)0, acc, false, false);
  }
  return acc;
}

// ---------------- weight conversion (f32 -> bf16, optional zero-pad) ----------------
__global__ void k_cvt_pad(bf16* dst, const float* src,
                          int rows_dst, int cols_dst, int rows_src, int cols_src) {
  int i = blockIdx.x * blockDim.x + threadIdx.x;
  int n = rows_dst * cols_dst;
  if (i >= n) return;
  int r = i / cols_dst, c = i - r * cols_dst;
  float v = (r < rows_src && c < cols_src) ? src[r * cols_src + c] : 0.f;
  dst[i] = f2bf(v);
}

// ---------------- encoder: embed + conv1d(k=5,pad=2) + relu -> xsB [S][16][512] bf16 ----------------
__global__ void k_embed_conv(const int* __restrict__ text, const float* __restrict__ emb,
                             const float* __restrict__ cw, const float* __restrict__ cb,
                             bf16* __restrict__ xsB) {
  int i = blockIdx.x * blockDim.x + threadIdx.x;   // 16*128*512
  int h = i & 511;
  int s = (i >> 9) & 127;
  int b = i >> 16;
  if (b >= B16) return;
  float acc = 0.f;
  if (b < 8) {
    acc = cb[h];
    for (int t = 0; t < 5; ++t) {
      int sp = s - 2 + t;
      if (sp < 0 || sp >= SLEN) continue;
      const float* er = emb + text[b * SLEN + sp] * 512;
      const float* wr = cw + h * 2560 + t;            // conv_w[h][e][t], stride 5 over e
      for (int e = 0; e < 512; ++e) acc += er[e] * wr[e * 5];
    }
    acc = fmaxf(acc, 0.f);
  }
  xsB[(s * B16 + b) * HDIM + h] = f2bf(acc);
}

// ---------------- encoder bi-LSTM: 2 blocks (dir), 256 thr (8 waves), 128 sequential steps ----------------
__global__ void k_enc_lstm(const bf16* __restrict__ xsB,
                           const bf16* wihF, const bf16* whhF, const float* bihF, const float* bhhF,
                           const bf16* wihR, const bf16* whhR, const float* bihR, const float* bhhR,
                           float* __restrict__ memF, bf16* __restrict__ memB) {
  extern __shared__ char smem[];
  float* cF = (float*)smem;                 // [16][256]
  bf16*  hB = (bf16*)(cF + B16 * 256);      // [16][256]
  const int dir = blockIdx.x;
  const bf16* wih = dir ? wihR : wihF;
  const bf16* whh = dir ? whhR : whhF;
  const float* bih = dir ? bihR : bihF;
  const float* bhh = dir ? bhhR : bhhF;
  const int tid = threadIdx.x, wave = tid >> 5, lane = tid & 31;
  const int half = lane >> 4, ln = lane & 15;

  for (int i = tid; i < B16 * 256; i += 256) { cF[i] = 0.f; hB[i] = f2bf(0.f); }
  __syncthreads();

  for (int step = 0; step < SLEN; ++step) {
    int s = dir ? (SLEN - 1 - step) : step;
    const bf16* Ax = xsB + s * B16 * HDIM;
    v8f acc[2][4];
#pragma unroll
    for (int jt = 0; jt < 2; ++jt) {
      int jb = (wave * 2 + jt) * 16;
#pragma unroll
      for (int q = 0; q < 4; ++q) {          // gate quarters i,f,g,o of 4*Hc=1024
        int nb = q * 256 + jb;
        v8f a = wmma_tile(Ax, HDIM, wih + nb * HDIM, HDIM, HDIM, vzero());
        a = wmma_tile(hB, 256, whh + nb * 256, 256, 256, a);
        float bb = bih[nb + ln] + bhh[nb + ln];
#pragma unroll
        for (int g = 0; g < 8; ++g) a[g] += bb;
        acc[jt][q] = a;
      }
    }
    __syncthreads();                          // all hB reads done before overwrite
#pragma unroll
    for (int jt = 0; jt < 2; ++jt) {
      int j = (wave * 2 + jt) * 16 + ln;
#pragma unroll
      for (int g = 0; g < 8; ++g) {
        int m = g + half * 8;
        float iv = sigm(acc[jt][0][g]);
        float fv = sigm(acc[jt][1][g]);
        float gv = tanhf(acc[jt][2][g]);
        float ov = sigm(acc[jt][3][g]);
        float c = fv * cF[m * 256 + j] + iv * gv;
        cF[m * 256 + j] = c;
        float h = ov * tanhf(c);
        hB[m * 256 + j] = f2bf(h);
        int col = dir * 256 + j;              // memory = concat(fwd, bwd)
        memF[(s * B16 + m) * HDIM + col] = h;
        memB[(s * B16 + m) * HDIM + col] = f2bf(h);
      }
    }
    __syncthreads();
  }
}

// ---------------- keys = memory @ attn_wk.T  (parallel, one tile per wave) ----------------
__global__ void k_keys(const bf16* __restrict__ memB, const bf16* __restrict__ wkB,
                       float* __restrict__ keysF) {
  int wave = threadIdx.x >> 5;
  int tile = blockIdx.x * 8 + wave;           // 128 * 32 tiles
  int s = tile >> 5, nb = (tile & 31) * 16;
  v8f a = wmma_tile(memB + s * B16 * HDIM, HDIM, wkB + nb * HDIM, HDIM, HDIM, vzero());
  int lane = threadIdx.x & 31, half = lane >> 4, ln = lane & 15;
#pragma unroll
  for (int g = 0; g < 8; ++g)
    keysF[(s * B16 + (g + half * 8)) * HDIM + nb + ln] = a[g];
}

// ---------------- decoder: 1 persistent block, 512 thr (16 waves), 200 sequential steps ----------------
__global__ void k_decoder(const float* __restrict__ memF, const float* __restrict__ keysF,
                          const bf16* wqB, const float* we,
                          const bf16* dwihB, const bf16* dwhhB, const float* dbih, const float* dbhh,
                          const bf16* pw1B, const float* pb1, const bf16* pw2B, const float* pb2,
                          const bf16* melwB, const float* melb, const bf16* stopwB, const float* stopb,
                          float* __restrict__ outMel, float* __restrict__ outStop) {
  extern __shared__ char smem[];
  float* cF    = (float*)smem;                // [16][512]
  float* qF    = cF + B16 * HDIM;             // [16][512]
  float* attnF = qF + B16 * HDIM;             // [16][128]
  bf16*  hB    = (bf16*)(attnF + B16 * SLEN); // [16][512]
  bf16*  p1B   = hB + B16 * HDIM;             // [16][512]
  bf16*  xinB  = p1B + B16 * HDIM;            // [16][1024]  (= [prenet|context] then [h2|context])
  bf16*  melB  = xinB + B16 * 1024;           // [16][96]    (K padded 80->96)

  const int tid = threadIdx.x;
  const int wave = tid >> 5, lane = tid & 31, half = lane >> 4, ln = lane & 15;

  for (int i = tid; i < B16 * HDIM; i += 512) { cF[i] = 0.f; hB[i] = f2bf(0.f); }
  for (int i = tid; i < B16 * 96; i += 512)   melB[i] = f2bf(0.f);
  __syncthreads();

  for (int t = 0; t < TSTEPS; ++t) {
    // -- prenet layer 1: relu(prev_mel @ pre_w1.T + b1) --
#pragma unroll
    for (int k = 0; k < 2; ++k) {
      int nb = (wave * 2 + k) * 16;
      v8f a = wmma_tile(melB, 96, pw1B + nb * 96, 96, 96, vzero());
      float bb = pb1[nb + ln];
#pragma unroll
      for (int g = 0; g < 8; ++g)
        p1B[(g + half * 8) * HDIM + nb + ln] = f2bf(fmaxf(a[g] + bb, 0.f));
    }
    __syncthreads();
    // -- prenet layer 2 (-> xin[:,0:512]) and query q = h @ wq.T --
#pragma unroll
    for (int k = 0; k < 2; ++k) {
      int nb = (wave * 2 + k) * 16;
      v8f a  = wmma_tile(p1B, HDIM, pw2B + nb * HDIM, HDIM, HDIM, vzero());
      v8f aq = wmma_tile(hB,  HDIM, wqB  + nb * HDIM, HDIM, HDIM, vzero());
      float bb = pb2[nb + ln];
#pragma unroll
      for (int g = 0; g < 8; ++g) {
        int m = g + half * 8;
        xinB[m * 1024 + nb + ln] = f2bf(fmaxf(a[g] + bb, 0.f));
        qF[m * HDIM + nb + ln]   = aq[g];
      }
    }
    __syncthreads();
    // -- additive attention energies: tanh(q + keys) . we --
    for (int e = tid; e < B16 * SLEN; e += 512) {
      int m = e >> 7, s = e & 127;
      const float* qr = qF + m * HDIM;
      const float* kr = keysF + (s * B16 + m) * HDIM;
      float sum = 0.f;
      for (int h2 = 0; h2 < HDIM; ++h2) sum += tanhf(qr[h2] + kr[h2]) * we[h2];
      attnF[e] = sum;
    }
    __syncthreads();
    // -- softmax over S (one wave per batch row, wave32 shuffles) --
    {
      int m = wave;
      float v0 = attnF[m * 128 + lane],      v1 = attnF[m * 128 + lane + 32];
      float v2 = attnF[m * 128 + lane + 64], v3 = attnF[m * 128 + lane + 96];
      float mx = fmaxf(fmaxf(v0, v1), fmaxf(v2, v3));
      for (int d = 16; d; d >>= 1) mx = fmaxf(mx, __shfl_xor(mx, d, 32));
      v0 = __expf(v0 - mx); v1 = __expf(v1 - mx);
      v2 = __expf(v2 - mx); v3 = __expf(v3 - mx);
      float sm = v0 + v1 + v2 + v3;
      for (int d = 16; d; d >>= 1) sm += __shfl_xor(sm, d, 32);
      float inv = 1.0f / sm;
      attnF[m * 128 + lane]      = v0 * inv; attnF[m * 128 + lane + 32] = v1 * inv;
      attnF[m * 128 + lane + 64] = v2 * inv; attnF[m * 128 + lane + 96] = v3 * inv;
    }
    __syncthreads();
    // -- context = attn @ memory (-> xin[:,512:1024]) --
    for (int e = tid; e < B16 * HDIM; e += 512) {
      int m = e >> 9, h2 = e & 511;
      const float* ar = attnF + m * 128;
      float sum = 0.f;
      for (int s = 0; s < SLEN; ++s) sum += ar[s] * memF[(s * B16 + m) * HDIM + h2];
      xinB[m * 1024 + 512 + h2] = f2bf(sum);
    }
    __syncthreads();
    // -- decoder LSTM gates: each wave owns (i,f,g,o) quadruple for its 32 hidden units --
    v8f acc[2][4];
#pragma unroll
    for (int jt = 0; jt < 2; ++jt) {
      int jb = (wave * 2 + jt) * 16;
#pragma unroll
      for (int q = 0; q < 4; ++q) {
        int nb = q * 512 + jb;                 // quarters of 4H=2048
        v8f a = wmma_tile(xinB, 1024, dwihB + nb * 1024, 1024, 1024, vzero());
        a = wmma_tile(hB, HDIM, dwhhB + nb * HDIM, HDIM, HDIM, a);
        float bb = dbih[nb + ln] + dbhh[nb + ln];
#pragma unroll
        for (int g = 0; g < 8; ++g) a[g] += bb;
        acc[jt][q] = a;
      }
    }
    __syncthreads();                           // all xin/hB reads done before overwrite
    // -- per-lane cell update; co = [h2, context] reuses xinB --
#pragma unroll
    for (int jt = 0; jt < 2; ++jt) {
      int j = (wave * 2 + jt) * 16 + ln;
#pragma unroll
      for (int g = 0; g < 8; ++g) {
        int m = g + half * 8;
        float iv = sigm(acc[jt][0][g]);
        float fv = sigm(acc[jt][1][g]);
        float gv = tanhf(acc[jt][2][g]);
        float ov = sigm(acc[jt][3][g]);
        float c = fv * cF[m * HDIM + j] + iv * gv;
        cF[m * HDIM + j] = c;
        float h = ov * tanhf(c);
        hB[m * HDIM + j]  = f2bf(h);
        xinB[m * 1024 + j] = f2bf(h);
      }
    }
    __syncthreads();
    // -- output heads: mel (5 tiles) and stop (1 padded tile) --
    if (wave < 5) {
      int nb = wave * 16;
      v8f a = wmma_tile(xinB, 1024, melwB + nb * 1024, 1024, 1024, vzero());
      int n = nb + ln;
      if (n < NMELS) {
        float bb = melb[n];
#pragma unroll
        for (int g = 0; g < 8; ++g) {
          int m = g + half * 8;
          float v = a[g] + bb;
          melB[m * 96 + n] = f2bf(v);          // prev_mel for next step
          if (m < 8) outMel[(m * TSTEPS + t) * NMELS + n] = v;
        }
      }
    } else if (wave == 5) {
      v8f a = wmma_tile(xinB, 1024, stopwB, 1024, 1024, vzero());
      if (ln == 0) {
        float b0 = stopb[0];
#pragma unroll
        for (int g = 0; g < 8; ++g) {
          int m = g + half * 8;
          if (m < 8) outStop[m * TSTEPS + t] = sigm(a[g] + b0);
        }
      }
    }
    __syncthreads();
  }
}

// ---------------- host launcher ----------------
extern "C" void kernel_launch(void* const* d_in, const int* in_sizes, int n_in,
                              void* d_out, int out_size, void* d_ws, size_t ws_size,
                              hipStream_t stream) {
  (void)in_sizes; (void)n_in; (void)out_size; (void)ws_size;
  const int*   text   = (const int*)d_in[0];
  const float* emb    = (const float*)d_in[3];
  const float* conv_w = (const float*)d_in[4];
  const float* conv_b = (const float*)d_in[5];
  const float* ewihf  = (const float*)d_in[6];
  const float* ewhhf  = (const float*)d_in[7];
  const float* ebihf  = (const float*)d_in[8];
  const float* ebhhf  = (const float*)d_in[9];
  const float* ewihb  = (const float*)d_in[10];
  const float* ewhhb  = (const float*)d_in[11];
  const float* ebihb  = (const float*)d_in[12];
  const float* ebhhb  = (const float*)d_in[13];
  const float* wq     = (const float*)d_in[14];
  const float* wk     = (const float*)d_in[15];
  const float* we     = (const float*)d_in[16];
  const float* dwih   = (const float*)d_in[17];
  const float* dwhh   = (const float*)d_in[18];
  const float* dbih   = (const float*)d_in[19];
  const float* dbhh   = (const float*)d_in[20];
  const float* pw1    = (const float*)d_in[21];
  const float* pb1    = (const float*)d_in[22];
  const float* pw2    = (const float*)d_in[23];
  const float* pb2    = (const float*)d_in[24];
  const float* melw   = (const float*)d_in[25];
  const float* melb   = (const float*)d_in[26];
  const float* stopw  = (const float*)d_in[27];
  const float* stopb  = (const float*)d_in[28];

  char* ws = (char*)d_ws;
  size_t off = 0;
  auto take = [&](size_t bytes) -> void* {
    void* p = ws + off;
    off = (off + bytes + 255) & ~(size_t)255;
    return p;
  };
  bf16*  xsB    = (bf16*)take((size_t)SLEN * B16 * HDIM * 2);
  float* memF   = (float*)take((size_t)SLEN * B16 * HDIM * 4);
  bf16*  memB   = (bf16*)take((size_t)SLEN * B16 * HDIM * 2);
  float* keysF  = (float*)take((size_t)SLEN * B16 * HDIM * 4);
  bf16*  wihfB  = (bf16*)take((size_t)1024 * 512 * 2);
  bf16*  whhfB  = (bf16*)take((size_t)1024 * 256 * 2);
  bf16*  wihbB  = (bf16*)take((size_t)1024 * 512 * 2);
  bf16*  whhbB  = (bf16*)take((size_t)1024 * 256 * 2);
  bf16*  wqB    = (bf16*)take((size_t)512 * 512 * 2);
  bf16*  wkB    = (bf16*)take((size_t)512 * 512 * 2);
  bf16*  dwihB  = (bf16*)take((size_t)2048 * 1024 * 2);
  bf16*  dwhhB  = (bf16*)take((size_t)2048 * 512 * 2);
  bf16*  pw1B   = (bf16*)take((size_t)512 * 96 * 2);
  bf16*  pw2B   = (bf16*)take((size_t)512 * 512 * 2);
  bf16*  melwB  = (bf16*)take((size_t)NMELS * 1024 * 2);
  bf16*  stopwB = (bf16*)take((size_t)16 * 1024 * 2);

  auto cvt = [&](bf16* dst, const float* src, int rd, int cd, int rs, int cs) {
    int n = rd * cd;
    k_cvt_pad<<<(n + 255) / 256, 256, 0, stream>>>(dst, src, rd, cd, rs, cs);
  };
  cvt(wihfB, ewihf, 1024, 512, 1024, 512);
  cvt(whhfB, ewhhf, 1024, 256, 1024, 256);
  cvt(wihbB, ewihb, 1024, 512, 1024, 512);
  cvt(whhbB, ewhhb, 1024, 256, 1024, 256);
  cvt(wqB,   wq,    512, 512, 512, 512);
  cvt(wkB,   wk,    512, 512, 512, 512);
  cvt(dwihB, dwih,  2048, 1024, 2048, 1024);
  cvt(dwhhB, dwhh,  2048, 512, 2048, 512);
  cvt(pw1B,  pw1,   512, 96, 512, NMELS);     // K padded 80 -> 96
  cvt(pw2B,  pw2,   512, 512, 512, 512);
  cvt(melwB, melw,  NMELS, 1024, NMELS, 1024);
  cvt(stopwB, stopw, 16, 1024, 1, 1024);       // N padded 1 -> 16

  k_embed_conv<<<(B16 * SLEN * HDIM) / 256, 256, 0, stream>>>(text, emb, conv_w, conv_b, xsB);

  size_t lstm_smem = (size_t)B16 * 256 * 4 + (size_t)B16 * 256 * 2;   // 24 KB
  k_enc_lstm<<<2, 256, lstm_smem, stream>>>(xsB,
      wihfB, whhfB, ebihf, ebhhf,
      wihbB, whhbB, ebihb, ebhhb,
      memF, memB);

  k_keys<<<(SLEN * 32) / 8, 256, 0, stream>>>(memB, wkB, keysF);

  float* outMel  = (float*)d_out;
  float* outStop = outMel + 8 * TSTEPS * NMELS;
  size_t dec_smem = (size_t)B16 * HDIM * 4      // cF
                  + (size_t)B16 * HDIM * 4      // qF
                  + (size_t)B16 * SLEN * 4      // attnF
                  + (size_t)B16 * HDIM * 2      // hB
                  + (size_t)B16 * HDIM * 2      // p1B
                  + (size_t)B16 * 1024 * 2      // xinB
                  + (size_t)B16 * 96 * 2;       // melB  (~139 KB, fits 320 KB WGP LDS)
  k_decoder<<<1, 512, dec_smem, stream>>>(memF, keysF,
      wqB, we,
      dwihB, dwhhB, dbih, dbhh,
      pw1B, pb1, pw2B, pb2,
      melwB, melb, stopwB, stopb,
      outMel, outStop);
}
